// SAGE_61967788146768
// MI455X (gfx1250) — matrix operand order repaired and checked by reading, not compiled, not run on hardware
//
#include <hip/hip_runtime.h>
#include <hip/hip_bf16.h>

typedef __attribute__((ext_vector_type(16))) __bf16 v16bf;
typedef __attribute__((ext_vector_type(8)))  __bf16 v8bf;
typedef __attribute__((ext_vector_type(4)))  __bf16 v4bf;
typedef __attribute__((ext_vector_type(8)))  float  v8f;

#define N_NODES 50000
#define N_EDGES 800000
// All layers: K = 512 (256 self feats ++ 256 mean feats), 16 k-steps of 32.

// ---------------- degree kernels ----------------
__global__ void k_deg(const int* __restrict__ dst, float* __restrict__ deg, int E) {
    int i = blockIdx.x * blockDim.x + threadIdx.x;
    if (i < E) atomicAdd(&deg[dst[i]], 1.0f);
}

__global__ void k_deg_inv(float* __restrict__ deg, int n) {
    int i = blockIdx.x * blockDim.x + threadIdx.x;
    if (i < n) deg[i] = 1.0f / fmaxf(deg[i], 1.0f);
}

// ------------- edge scatter-add (segment_sum), F=256 -------------
// one wave per edge: indices loaded once (wave-uniform), 32 lanes cover 256 feats
__global__ void __launch_bounds__(256)
k_scatter(const float* __restrict__ h, const int* __restrict__ src,
          const int* __restrict__ dst, float* __restrict__ agg, int E) {
    int wave = (int)((blockIdx.x * (long long)blockDim.x + threadIdx.x) >> 5);
    int lane = threadIdx.x & 31;
    if (wave >= E) return;
    int s = src[wave];
    int d = dst[wave];
    const float4* hr = (const float4*)(h + (size_t)s * 256);
    float* ar = agg + (size_t)d * 256;
#pragma unroll
    for (int i = 0; i < 2; ++i) {
        int c4 = lane + i * 32;                 // float4 chunk 0..63
        float4 v = hr[c4];
        float* a = ar + c4 * 4;
        atomicAdd(a + 0, v.x);
        atomicAdd(a + 1, v.y);
        atomicAdd(a + 2, v.z);
        atomicAdd(a + 3, v.w);
    }
}

// ------------- build bf16 Xcat[N][512] = [ h | agg*deg_inv ] -------------
__global__ void k_pack_x(const float* __restrict__ h, const float* __restrict__ agg,
                         const float* __restrict__ deg_inv, __bf16* __restrict__ xc, int n) {
    long long idx = (long long)blockIdx.x * blockDim.x + threadIdx.x;
    int node = (int)(idx >> 6);
    int f4   = (int)(idx & 63);
    if (node >= n) return;
    float4 hv = ((const float4*)(h + (size_t)node * 256))[f4];
    float4 av = ((const float4*)(agg + (size_t)node * 256))[f4];
    float di  = deg_inv[node];
    v4bf hb, mb;
    hb[0] = (__bf16)hv.x; hb[1] = (__bf16)hv.y; hb[2] = (__bf16)hv.z; hb[3] = (__bf16)hv.w;
    mb[0] = (__bf16)(av.x * di); mb[1] = (__bf16)(av.y * di);
    mb[2] = (__bf16)(av.z * di); mb[3] = (__bf16)(av.w * di);
    ((v4bf*)(xc + (size_t)node * 512))[f4]       = hb;
    ((v4bf*)(xc + (size_t)node * 512 + 256))[f4] = mb;
}

// ------------- permute stacked weights into B-fragment layout -------------
// Wcat[K=512][Fout], K<256 -> W_self, K>=256 -> W_neigh.
// wp[((ks*NT + nt)*32 + lane)*16 + j] = bf16(Wcat[ks*32 + koff + (j&7) + (j>>3)*16][nt*16 + lane%16])
// koff = lane<16 ? 0 : 8   (lanes 0-15 K=0..7,16..23; lanes 16-31 K=8..15,24..31)
__global__ void k_pack_w(const float* __restrict__ Wself, const float* __restrict__ Wneigh,
                         __bf16* __restrict__ wp, int Fout) {
    int t = blockIdx.x * blockDim.x + threadIdx.x;
    int NT = Fout >> 4;
    if (t >= 16 * NT * 32) return;
    int lane = t & 31;
    int rest = t >> 5;
    int nt = rest % NT;
    int ks = rest / NT;
    int n  = nt * 16 + (lane & 15);
    int koff = (lane < 16) ? 0 : 8;
    __bf16* o = wp + (size_t)t * 16;
#pragma unroll
    for (int j = 0; j < 16; ++j) {
        int k = ks * 32 + koff + (j & 7) + ((j >> 3) << 4);
        float w = (k < 256) ? Wself[(size_t)k * Fout + n]
                            : Wneigh[(size_t)(k - 256) * Fout + n];
        o[j] = (__bf16)w;
    }
}

// ------------- WMMA GEMM: out[M x Fout] = Xcat[M x 512] @ Wcat + bias (opt ReLU) -------
// Each wave computes a 16x64 strip: one A fragment reused across 4 N-tiles.
__global__ void __launch_bounds__(256)
k_gemm(const __bf16* __restrict__ X, const __bf16* __restrict__ Wp,
       const float* __restrict__ bias, float* __restrict__ out,
       int Mt, int NT, int relu) {
    int wave = (int)((blockIdx.x * (long long)blockDim.x + threadIdx.x) >> 5);
    int lane = threadIdx.x & 31;
    int NG = NT >> 2;                           // N-tile groups of 4
    if (wave >= Mt * NG) return;                // wave-uniform; EXEC stays all-1s
    int mt  = wave / NG;
    int nt0 = (wave % NG) * 4;

    // A fragment source: row-major bf16 row, two contiguous 16B loads per lane per k-step
    int row = mt * 16 + (lane & 15);
    int klo = (lane < 16) ? 0 : 8;
    const __bf16* xr = X + (size_t)row * 512 + klo;

    v8f c0 = {}, c1 = {}, c2 = {}, c3 = {};
#pragma unroll
    for (int ks = 0; ks < 16; ++ks) {
        v8bf alo = *(const v8bf*)(xr + ks * 32);        // K = klo .. klo+7
        v8bf ahi = *(const v8bf*)(xr + ks * 32 + 16);   // K = klo+16 .. klo+23
        v16bf a;
#pragma unroll
        for (int j = 0; j < 8; ++j) { a[j] = alo[j]; a[j + 8] = ahi[j]; }
        const __bf16* wb = Wp + (((size_t)ks * NT + nt0) * 32 + lane) * 16;
        v16bf b0 = *(const v16bf*)(wb);
        v16bf b1 = *(const v16bf*)(wb + 32 * 16);
        v16bf b2 = *(const v16bf*)(wb + 64 * 16);
        v16bf b3 = *(const v16bf*)(wb + 96 * 16);
        c0 = __builtin_amdgcn_wmma_f32_16x16x32_bf16(false, a, false, b0, (short)0, c0, false, false);
        c1 = __builtin_amdgcn_wmma_f32_16x16x32_bf16(false, a, false, b1, (short)0, c1, false, false);
        c2 = __builtin_amdgcn_wmma_f32_16x16x32_bf16(false, a, false, b2, (short)0, c2, false, false);
        c3 = __builtin_amdgcn_wmma_f32_16x16x32_bf16(false, a, false, b3, (short)0, c3, false, false);
    }

    // C/D layout: lane -> N column, VGPR r -> row (lanes 16-31 offset by 8)
    int Fout = NT * 16;
    int mbase = mt * 16 + ((lane < 16) ? 0 : 8);
    v8f acc[4] = {c0, c1, c2, c3};
#pragma unroll
    for (int u = 0; u < 4; ++u) {
        int n = (nt0 + u) * 16 + (lane & 15);
        float bv = bias[n];
#pragma unroll
        for (int r = 0; r < 8; ++r) {
            float v = acc[u][r] + bv;
            if (relu) v = fmaxf(v, 0.0f);
            out[(size_t)(mbase + r) * Fout + n] = v;
        }
    }
}

extern "C" void kernel_launch(void* const* d_in, const int* in_sizes, int n_in,
                              void* d_out, int out_size, void* d_ws, size_t ws_size,
                              hipStream_t stream) {
    const float* x   = (const float*)d_in[0];
    const int*   src = (const int*)d_in[1];
    const int*   dst = (const int*)d_in[2];
    const float* Ws1 = (const float*)d_in[3];
    const float* Wn1 = (const float*)d_in[4];
    const float* b1  = (const float*)d_in[5];
    const float* Ws2 = (const float*)d_in[6];
    const float* Wn2 = (const float*)d_in[7];
    const float* b2  = (const float*)d_in[8];
    const float* Ws3 = (const float*)d_in[9];
    const float* Wn3 = (const float*)d_in[10];
    const float* b3  = (const float*)d_in[11];

    char* ws = (char*)d_ws;
    size_t o = 0;
    auto alloc = [&](size_t bytes) {
        size_t r = o;
        o += (bytes + 255) & ~(size_t)255;
        return r;
    };
    float*  deg = (float*)(ws + alloc((size_t)N_NODES * 4));
    float*  agg = (float*)(ws + alloc((size_t)N_NODES * 256 * 4));
    float*  h1  = (float*)(ws + alloc((size_t)N_NODES * 256 * 4));
    float*  h2  = (float*)(ws + alloc((size_t)N_NODES * 256 * 4));
    __bf16* xc  = (__bf16*)(ws + alloc((size_t)N_NODES * 512 * 2));
    __bf16* wp  = (__bf16*)(ws + alloc((size_t)512 * 256 * 2));

    const int MT = N_NODES / 16;                              // 3125 row tiles (exact)
    const int sc_blocks = (int)(((long long)N_EDGES * 32 + 255) / 256);
    const int px_blocks = (int)(((long long)N_NODES * 64 + 255) / 256);
    const int g256_blocks = (MT * 4 + 7) / 8;                 // Fout=256: 4 N-groups/row-tile
    const int g128_blocks = (MT * 2 + 7) / 8;                 // Fout=128: 2 N-groups/row-tile

    // degrees (shared by all layers)
    hipMemsetAsync(deg, 0, (size_t)N_NODES * 4, stream);
    k_deg<<<(N_EDGES + 255) / 256, 256, 0, stream>>>(dst, deg, N_EDGES);
    k_deg_inv<<<(N_NODES + 255) / 256, 256, 0, stream>>>(deg, N_NODES);

    // ---- layer 1: x -> h1 (Fout=256, ReLU)
    hipMemsetAsync(agg, 0, (size_t)N_NODES * 256 * 4, stream);
    k_scatter<<<sc_blocks, 256, 0, stream>>>(x, src, dst, agg, N_EDGES);
    k_pack_x<<<px_blocks, 256, 0, stream>>>(x, agg, deg, xc, N_NODES);
    k_pack_w<<<(16 * 16 * 32 + 255) / 256, 256, 0, stream>>>(Ws1, Wn1, wp, 256);
    k_gemm<<<g256_blocks, 256, 0, stream>>>(xc, wp, b1, h1, MT, 16, 1);

    // ---- layer 2: h1 -> h2 (Fout=256, ReLU)
    hipMemsetAsync(agg, 0, (size_t)N_NODES * 256 * 4, stream);
    k_scatter<<<sc_blocks, 256, 0, stream>>>(h1, src, dst, agg, N_EDGES);
    k_pack_x<<<px_blocks, 256, 0, stream>>>(h1, agg, deg, xc, N_NODES);
    k_pack_w<<<(16 * 16 * 32 + 255) / 256, 256, 0, stream>>>(Ws2, Wn2, wp, 256);
    k_gemm<<<g256_blocks, 256, 0, stream>>>(xc, wp, b2, h2, MT, 16, 1);

    // ---- layer 3: h2 -> d_out (Fout=128, no ReLU)
    hipMemsetAsync(agg, 0, (size_t)N_NODES * 256 * 4, stream);
    k_scatter<<<sc_blocks, 256, 0, stream>>>(h2, src, dst, agg, N_EDGES);
    k_pack_x<<<px_blocks, 256, 0, stream>>>(h2, agg, deg, xc, N_NODES);
    k_pack_w<<<(16 * 8 * 32 + 255) / 256, 256, 0, stream>>>(Ws3, Wn3, wp, 128);
    k_gemm<<<g128_blocks, 256, 0, stream>>>(xc, wp, b3, (float*)d_out, MT, 8, 0);
}
